// Classifier_custom_12936441496172
// MI455X (gfx1250) — compile-verified, matrix operand order checked by reading, not run on hardware
//
#include <hip/hip_runtime.h>

typedef __attribute__((ext_vector_type(16))) _Float16 v16h;
typedef __attribute__((ext_vector_type(8)))  _Float16 v8h;
typedef __attribute__((ext_vector_type(4)))  _Float16 v4h;
typedef __attribute__((ext_vector_type(8)))  float    v8f;

#define N_ATT      312
#define N_ATT_PAD  320
#define DV         300
#define CF         1024
#define NB         128
#define RR         256
#define CHUNK_K    32
#define NCHUNK     (CF / CHUNK_K)      // 32
#define LDS_STRIDE 36                  // halfs per r-row (72B: 8B-aligned, conflict-free)
#define ROWS_WG    64
#define A_STRIDE   257                 // floats per row of A tile
#define G_STRIDE   260                 // halfs per row of G tile

// ---------------- prep kernels ----------------

__global__ void norm_v_kernel(const float* __restrict__ V, float* __restrict__ invnV) {
    int i = threadIdx.x;
    if (i >= N_ATT_PAD) return;
    float s = 0.f;
    if (i < N_ATT) {
        const float* p = V + (size_t)i * DV;
        for (int v = 0; v < DV; ++v) s += p[v] * p[v];
    }
    invnV[i] = 1.f / fmaxf(sqrtf(s), 1e-12f);
}

__global__ __launch_bounds__(256)
void qproj_kernel(const float* __restrict__ V, const float* __restrict__ W1,
                  const float* __restrict__ W2, const float* __restrict__ invnV,
                  _Float16* __restrict__ Qh, _Float16* __restrict__ Q2h) {
    int f = blockIdx.x * 256 + threadIdx.x;
    int i = blockIdx.y;
    float q1 = 0.f, q2 = 0.f;
    if (i < N_ATT) {
        float inv = invnV[i];
        const float* vr = V + (size_t)i * DV;
        for (int v = 0; v < DV; ++v) {
            float vv = vr[v] * inv;
            q1 += vv * W1[(size_t)v * CF + f];
            q2 += vv * W2[(size_t)v * CF + f];
        }
    }
    Qh [(size_t)i * CF + f] = (_Float16)q1;
    Q2h[(size_t)i * CF + f] = (_Float16)q2;
}

__global__ __launch_bounds__(256)
void norm_img_kernel(const float* __restrict__ img, float* __restrict__ invnI) {
    int b = blockIdx.x, r = threadIdx.x;
    const float* p = img + (size_t)b * CF * RR + r;
    float s = 0.f;
    for (int f = 0; f < CF; ++f) { float x = p[(size_t)f * RR]; s += x * x; }
    invnI[b * RR + r] = 1.f / fmaxf(sqrtf(s), 1e-12f);
}

// ---------------- fused attention kernel ----------------
// grid (5, 128), block 256 = 8 waves. Wave w: i-rows (w&3)*16.., r-tiles (w>>2)*8..+7.
// Double-buffered LDS chunk + register double-buffered B fragments.
__global__ __launch_bounds__(256)
void attn_fused_kernel(const float* __restrict__ img, const _Float16* __restrict__ Qh,
                       const _Float16* __restrict__ Q2h, const float* __restrict__ invn,
                       float* __restrict__ out) {
    __shared__ _Float16 chunk[2][RR * LDS_STRIDE];        // av chunks        (36 KB)
    __shared__ float    A_ls[ROWS_WG * A_STRIDE];         // logits f32       (64.3 KB)
    __shared__ _Float16 G_ls[ROWS_WG * G_STRIDE];         // G tile f16       (32.5 KB)
    __shared__ float    red[2 * 256];                     // reductions       (2 KB)

    const int t      = threadIdx.x;
    const int lane   = t & 31;
    const int wave   = t >> 5;
    const int b      = blockIdx.y;
    const int i0     = blockIdx.x * ROWS_WG;
    const int rowgrp = (wave & 3) * 16;
    const int rtbase = (wave >> 2) * 8;

    const int m  = lane & 15;
    const int kb = (lane >= 16) ? 8 : 0;

    // chunk staging: one r-row per thread
    const int   r    = t;
    const float invr = invn[b * RR + r];
    const float* imgB = img + (size_t)b * CF * RR;

    const _Float16* qrow  = Qh  + (size_t)(i0 + rowgrp + m) * CF + kb;
    const _Float16* q2row = Q2h + (size_t)(i0 + rowgrp + m) * CF + kb;

    v8f accA[8], accG[8];
#pragma unroll
    for (int rt = 0; rt < 8; ++rt) { accA[rt] = (v8f)0.f; accG[rt] = (v8f)0.f; }

    float stage[CHUNK_K];

    auto load_stage = [&](int fc) {
        const float* src = imgB + (size_t)fc * CHUNK_K * RR + r;
#pragma unroll
        for (int fl = 0; fl < CHUNK_K; ++fl) stage[fl] = src[(size_t)fl * RR];
    };
    auto store_stage = [&](int bi) {
        _Float16* dst = &chunk[bi][r * LDS_STRIDE];
#pragma unroll
        for (int j = 0; j < CHUNK_K / 4; ++j) {
            v4h h;
#pragma unroll
            for (int k = 0; k < 4; ++k) h[k] = (_Float16)(stage[4 * j + k] * invr);
            *(v4h*)(dst + 4 * j) = h;
        }
    };
    auto load_afrag = [&](int fc, v16h& A, v16h& G) {
        v8h alo = *(const v8h*)(qrow  + fc * CHUNK_K);
        v8h ahi = *(const v8h*)(qrow  + fc * CHUNK_K + 16);
        v8h glo = *(const v8h*)(q2row + fc * CHUNK_K);
        v8h ghi = *(const v8h*)(q2row + fc * CHUNK_K + 16);
#pragma unroll
        for (int j = 0; j < 8; ++j) {
            A[j] = alo[j]; A[8 + j] = ahi[j];
            G[j] = glo[j]; G[8 + j] = ghi[j];
        }
    };
    auto load_bfrag = [&](const _Float16* cbase, int rt) {
        const _Float16* rp = cbase + (rt * 16 + m) * LDS_STRIDE + kb;
        v4h p0 = *(const v4h*)(rp);
        v4h p1 = *(const v4h*)(rp + 4);
        v4h p2 = *(const v4h*)(rp + 16);
        v4h p3 = *(const v4h*)(rp + 20);
        v16h bf;
#pragma unroll
        for (int j = 0; j < 4; ++j) {
            bf[j] = p0[j]; bf[4 + j] = p1[j]; bf[8 + j] = p2[j]; bf[12 + j] = p3[j];
        }
        return bf;
    };

    // prologue: chunk 0 into buffer 0, first A fragments
    v16h aQ, aG;
    load_stage(0);
    load_afrag(0, aQ, aG);
    store_stage(0);
    __syncthreads();

    for (int fc = 0; fc < NCHUNK; ++fc) {
        const int last = (fc == NCHUNK - 1);
        v16h aQn, aGn;
        if (!last) {
            load_stage(fc + 1);            // global loads in flight over WMMA block
            load_afrag(fc + 1, aQn, aGn);
        }

        const _Float16* cbase = chunk[fc & 1];
        v16h bf = load_bfrag(cbase, rtbase);
#pragma unroll
        for (int rtl = 0; rtl < 8; ++rtl) {
            v16h bfn = (rtl < 7) ? load_bfrag(cbase, rtbase + rtl + 1) : bf;
            accA[rtl] = __builtin_amdgcn_wmma_f32_16x16x32_f16(
                false, aQ, false, bf, (short)0, accA[rtl], false, false);
            accG[rtl] = __builtin_amdgcn_wmma_f32_16x16x32_f16(
                false, aG, false, bf, (short)0, accG[rtl], false, false);
            bf = bfn;
        }

        if (!last) {
            store_stage((fc + 1) & 1);
            aQ = aQn; aG = aGn;
        }
        __syncthreads();
    }

    // dump accumulators (16x16 f32 C/D layout: VGPR j -> row j or j+8)
    {
        const int rowbase = rowgrp + ((lane < 16) ? 0 : 8);
#pragma unroll
        for (int rtl = 0; rtl < 8; ++rtl) {
            const int colt = (rtbase + rtl) * 16 + m;
#pragma unroll
            for (int j = 0; j < 8; ++j) {
                A_ls[(rowbase + j) * A_STRIDE + colt] = accA[rtl][j];
                G_ls[(rowbase + j) * G_STRIDE + colt] = (_Float16)accG[rtl][j];
            }
        }
    }
    __syncthreads();

    // fused softmax + weighted reduce: 4 threads per row, 64 cols each
    const int row = t >> 2;
    const int q   = t & 3;
    const float*    Arow = &A_ls[row * A_STRIDE + q * 64];
    const _Float16* Grow = &G_ls[row * G_STRIDE + q * 64];

    float lm = -1e30f;
    for (int c = 0; c < 64; ++c) lm = fmaxf(lm, Arow[c]);
    red[t] = lm;
    __syncthreads();
    const float mg = fmaxf(fmaxf(red[row * 4], red[row * 4 + 1]),
                           fmaxf(red[row * 4 + 2], red[row * 4 + 3]));

    float s = 0.f, tg = 0.f;
    for (int c = 0; c < 64; ++c) {
        float e = __expf(Arow[c] - mg);
        s  += e;
        tg += e * (float)Grow[c];
    }
    __syncthreads();
    red[t]       = s;
    red[256 + t] = tg;
    __syncthreads();
    if (q == 0) {
        const int i = i0 + row;
        if (i < N_ATT) {
            float stot = red[row * 4] + red[row * 4 + 1] + red[row * 4 + 2] + red[row * 4 + 3];
            float ttot = red[256 + row * 4] + red[256 + row * 4 + 1] +
                         red[256 + row * 4 + 2] + red[256 + row * 4 + 3];
            out[(size_t)b * N_ATT + i] = ttot / stot;
        }
    }
}

// ---------------- launcher ----------------

extern "C" void kernel_launch(void* const* d_in, const int* in_sizes, int n_in,
                              void* d_out, int out_size, void* d_ws, size_t ws_size,
                              hipStream_t stream) {
    const float* img = (const float*)d_in[0];   // [128,1024,16,16]
    const float* V   = (const float*)d_in[1];   // [312,300]
    const float* W1  = (const float*)d_in[2];   // [300,1024]
    const float* W2  = (const float*)d_in[3];   // [300,1024]
    float* out = (float*)d_out;                 // [128,312]

    char* ws = (char*)d_ws;
    float*    invnV = (float*)ws;                              // 320 floats
    float*    invnI = invnV + N_ATT_PAD;                       // 128*256 floats
    _Float16* Qh    = (_Float16*)(invnI + NB * RR);            // 320*1024 halfs
    _Float16* Q2h   = Qh + (size_t)N_ATT_PAD * CF;             // 320*1024 halfs

    norm_v_kernel  <<<1, N_ATT_PAD, 0, stream>>>(V, invnV);
    qproj_kernel   <<<dim3(CF / 256, N_ATT_PAD), 256, 0, stream>>>(V, W1, W2, invnV, Qh, Q2h);
    norm_img_kernel<<<NB, RR, 0, stream>>>(img, invnI);
    attn_fused_kernel<<<dim3(N_ATT_PAD / ROWS_WG, NB), 256, 0, stream>>>(img, Qh, Q2h, invnI, out);
}
